// OverlappingCrossAttention_76776835383417
// MI455X (gfx1250) — matrix operand order repaired and checked by reading, not compiled, not run on hardware
//
#include <hip/hip_runtime.h>
#include <hip/hip_bf16.h>

// Overlapping cross attention, fused, f16 WMMA + TDM staging, one window/WG.
// B=4, C=192, H=W=256, WS=8, OV=4, OS=16, NH=6, hd=32.

typedef __attribute__((ext_vector_type(16))) _Float16 v16h;
typedef __attribute__((ext_vector_type(8)))  _Float16 v8h;
typedef __attribute__((ext_vector_type(2)))  _Float16 v2h;
typedef __attribute__((ext_vector_type(8)))  float    v8f;
typedef __attribute__((ext_vector_type(4)))  unsigned int v4u;
typedef __attribute__((ext_vector_type(8)))  int      v8i;
typedef __attribute__((ext_vector_type(4)))  int      v4i;

__device__ __forceinline__ v8f wmma_f16(v16h a, v16h b, v8f c) {
  // D = A(16x32 f16) * B(32x16 f16) + C(16x16 f32)
  return __builtin_amdgcn_wmma_f32_16x16x32_f16(
      /*neg_a=*/false, a, /*neg_b=*/false, b,
      /*c_mod=*/(short)0, c, /*reuse_a=*/false, /*reuse_b=*/false);
}

// A fragment: lane (m = lane&15, sel = lane>>4) holds K chunk
// [kbase+sel*16 .. +16). Caller passes
// p = rowbase + kbase + sel*8 ; chunk starts at p+sel*8 elements.
__device__ __forceinline__ v16h mkA(const _Float16* p) {
  const int sl8 = ((threadIdx.x & 31) >> 4) * 8;
  v8h lo = *(const v8h*)(p + sl8);
  v8h hi = *(const v8h*)(p + sl8 + 8);
  v16h r;
#pragma unroll
  for (int i = 0; i < 8; ++i) { r[i] = lo[i]; r[8 + i] = hi[i]; }
  return r;
}

// ---- weight pre-transpose + f16 convert: WT[out][in] = W[in][out] ----
__global__ void prep_weights(const float* __restrict__ Wq,
                             const float* __restrict__ Wkv,
                             const float* __restrict__ Wo,
                             _Float16* __restrict__ WqT,
                             _Float16* __restrict__ WkvT,
                             _Float16* __restrict__ WoT) {
  int i = blockIdx.x * 256 + threadIdx.x;
  if (i < 192 * 192) {
    int o = i / 192, k = i % 192;
    WqT[i] = (_Float16)Wq[k * 192 + o];
    WoT[i] = (_Float16)Wo[k * 192 + o];
  }
  if (i < 384 * 192) {
    int o = i / 192, k = i % 192;
    WkvT[i] = (_Float16)Wkv[k * 384 + o];
  }
}

// LDS layout (bytes), total 323584:
//   [0      , 98304 ) ln   f16 [256][192]   -- later aliased:
//                       Sbuf f32 [64][256] @0, Pbuf f16 [64][256] @65536,
//                       outb f32 [64][192] @0 (final)
//   [98304  , 196608) K    f16 [256][192]   \  staging f32 [192][256]
//   [196608 , 294912) Vt   f16 [192][256]   /  (c-major, TDM fill order)
//   [294912 , 319488) q    f16 [64][192]
//   [319488 , 323584) Obuf f16 [64][32]
__global__ __launch_bounds__(256) void oca_main(
    const float* __restrict__ x, const float* __restrict__ gamma,
    const float* __restrict__ beta, const float* __restrict__ bq,
    const float* __restrict__ bkv, const float* __restrict__ bo,
    const _Float16* __restrict__ WqT, const _Float16* __restrict__ WkvT,
    const _Float16* __restrict__ WoT, float* __restrict__ out) {
  extern __shared__ char sm[];
  _Float16* lnp  = (_Float16*)sm;
  _Float16* Klds = (_Float16*)(sm + 98304);
  _Float16* Vt   = (_Float16*)(sm + 196608);
  _Float16* qlds = (_Float16*)(sm + 294912);
  _Float16* Obuf = (_Float16*)(sm + 319488);
  float*    Sbuf = (float*)sm;
  _Float16* Pbuf = (_Float16*)(sm + 65536);
  float*    stg  = (float*)(sm + 98304);   // [192][256] c-major staging
  float*    outb = (float*)sm;

  const int tid   = threadIdx.x;
  const int lane  = tid & 31;
  const int wave  = tid >> 5;
  const int lanem = lane & 15;
  const int sel   = lane >> 4;
  const int sel8  = sel * 8;

  const int bw = blockIdx.x;
  const int b  = bw >> 10;
  const int w  = bw & 1023;
  const int wy = w >> 5;
  const int wx = w & 31;

  // ---------------- Stage A: stage 16x16x192 patch into LDS ----------------
  {
    int pr = tid >> 4, pc = tid & 15;
    int gy = wy * 8 - 4 + pr, gx = wx * 8 - 4 + pc;
    bool inb = ((unsigned)gy < 256u) && ((unsigned)gx < 256u);
    bool interior = (wy >= 1) && (wy <= 30) && (wx >= 1) && (wx <= 30);

    if (interior) {
      // One Tensor-DMA descriptor moves the whole 196.6 KB tile -> LDS.
      if (wave == 0) {
        unsigned long long ga = (unsigned long long)(uintptr_t)(
            x + (size_t)b * 192 * 65536 + (size_t)(wy * 8 - 4) * 256 + (wx * 8 - 4));
        v4u g0;
        g0[0] = 1u;                                     // count=1, user mode
        g0[1] = 98304u;                                 // lds_addr (bytes)
        g0[2] = (unsigned)(ga & 0xFFFFFFFFu);           // global_addr[31:0]
        g0[3] = ((unsigned)(ga >> 32) & 0x01FFFFFFu)    // global_addr[56:32]
                | 0x80000000u;                          // type=2 (image)
        v8i g1;
        g1[0] = 0x00020000;            // wg_mask=0, data_size=4B
        g1[1] = (int)(256u << 16);     // tensor_dim0 = 256 (W)
        g1[2] = (int)(256u << 16);     // tensor_dim1 = 256 (H)
        g1[3] = (int)(16u << 16);      // tile_dim0 = 16
        g1[4] = (int)(16u | (192u << 16)); // tile_dim1=16, tile_dim2=192
        g1[5] = 256;                   // tensor_dim0_stride = W
        g1[6] = 0;                     // dim0_stride hi | dim1_stride lo16
        g1[7] = 1;                     // tensor_dim1_stride = 65536 (H*W)
        v4i g2 = {192, 0, 0, 0};       // tensor_dim2 = 192 (C)
        v4i g3 = {0, 0, 0, 0};
        v8i g4 = {0, 0, 0, 0, 0, 0, 0, 0};  // extra group (clang-23 form)
        __builtin_amdgcn_tensor_load_to_lds(g0, g1, g2, g3, g4, 0);
        __builtin_amdgcn_s_wait_tensorcnt((short)0);
      }
    } else if (inb) {
      // border window: per-lane async global->LDS copies (ASYNCcnt-tracked)
      unsigned lds0 = (unsigned)(uintptr_t)(&stg[tid]);
      const float* gp = x + (size_t)b * 192 * 65536 + (size_t)gy * 256 + gx;
      for (int c = 0; c < 192; ++c) {
        asm volatile("global_load_async_to_lds_b32 %0, %1, off"
                     :: "v"(lds0 + 1024u * c), "v"(gp) : "memory");
        gp += 65536;
      }
      asm volatile("s_wait_asynccnt 0x0" ::: "memory");
    } else {
      for (int c = 0; c < 192; ++c) stg[c * 256 + tid] = 0.f;
    }
    __syncthreads();

    // ---------------- LayerNorm over C per token -------------------------
    float s0 = 0.f, s1 = 0.f;
    for (int c = 0; c < 192; ++c) {
      float v = stg[c * 256 + tid];
      s0 += v; s1 += v * v;
    }
    float mu  = s0 * (1.f / 192.f);
    float var = s1 * (1.f / 192.f) - mu * mu;
    float rs  = rsqrtf(var + 1e-5f);
    for (int c = 0; c < 192; c += 2) {
      float v0 = (stg[c * 256 + tid] - mu) * rs * gamma[c] + beta[c];
      float v1 = (stg[(c + 1) * 256 + tid] - mu) * rs * gamma[c + 1] + beta[c + 1];
      v2h pk;
      pk[0] = (_Float16)(inb ? v0 : 0.f);
      pk[1] = (_Float16)(inb ? v1 : 0.f);
      *(v2h*)(lnp + tid * 192 + c) = pk;  // zero-padded ln, token-major
    }
  }
  __syncthreads();

  const float scale = 0.17677669529663687f;  // 32^-0.5

  // ---------------- Stage B: kv = ln @ Wkv  (256x384, K=192) ---------------
  // Each wave owns 2 M-rows; A-fragments (6 K-steps) live in registers and
  // are reused across all 24 N-tiles -> minimal LDS traffic.
#pragma unroll
  for (int half = 0; half < 2; ++half) {
    int tm = wave + half * 8;  // 0..15
    v16h afr[6];
#pragma unroll
    for (int ks = 0; ks < 6; ++ks)
      afr[ks] = mkA(lnp + (tm * 16 + lanem) * 192 + ks * 32 + sel8);
    for (int tn = 0; tn < 24; ++tn) {
      v8f acc = {0, 0, 0, 0, 0, 0, 0, 0};
#pragma unroll
      for (int ks = 0; ks < 6; ++ks) {
        v16h bb = *(const v16h*)(WkvT + (tn * 16 + lanem) * 192 + ks * 32 + sel * 16);
        acc = wmma_f16(afr[ks], bb, acc);
      }
      int col = tn * 16 + lanem;
      float bias = bkv[col];
      if (tn < 12) {  // K half: [tok][192]
#pragma unroll
        for (int g = 0; g < 8; ++g)
          Klds[(tm * 16 + sel8 + g) * 192 + col] = (_Float16)(acc[g] + bias);
      } else {        // V half, transposed: [d][256]
        int vc = col - 192;
#pragma unroll
        for (int g = 0; g < 8; ++g)
          Vt[vc * 256 + tm * 16 + sel8 + g] = (_Float16)(acc[g] + bias);
      }
    }
  }

  // ---------------- Stage C: q = ln(center) @ Wq, scale folded in ----------
  // wave -> (tm = wave>>1, 6 N-tiles); A-fragments loaded once.
  {
    int tm  = wave >> 1;          // 0..3
    int tn0 = (wave & 1) * 6;     // 0 or 6
    int qr = tm * 16 + lanem;                     // query token 0..63
    int prow = (qr / 8 + 4) * 16 + (qr % 8 + 4);  // center of patch
    v16h afr[6];
#pragma unroll
    for (int ks = 0; ks < 6; ++ks)
      afr[ks] = mkA(lnp + prow * 192 + ks * 32 + sel8);
    for (int tn = tn0; tn < tn0 + 6; ++tn) {
      v8f acc = {0, 0, 0, 0, 0, 0, 0, 0};
#pragma unroll
      for (int ks = 0; ks < 6; ++ks) {
        v16h bb = *(const v16h*)(WqT + (tn * 16 + lanem) * 192 + ks * 32 + sel * 16);
        acc = wmma_f16(afr[ks], bb, acc);
      }
      int col = tn * 16 + lanem;
      float bias = bq[col];
#pragma unroll
      for (int g = 0; g < 8; ++g)
        qlds[(tm * 16 + sel8 + g) * 192 + col] = (_Float16)((acc[g] + bias) * scale);
    }
  }
  __syncthreads();

  // ---------------- Stage D: attention per head + incremental out-proj -----
  v8f oacc[6];
#pragma unroll
  for (int i = 0; i < 6; ++i) oacc[i] = (v8f){0, 0, 0, 0, 0, 0, 0, 0};

  for (int h = 0; h < 6; ++h) {
    // S = q . K^T  (64x256, K-dim = 32 -> one WMMA per tile)
    // wave -> (tm = wave&3, 8 K-tiles); single A-fragment reused 8x.
    {
      int tm  = wave & 3;
      int tn0 = (wave >> 2) * 8;
      v16h a = mkA(qlds + (tm * 16 + lanem) * 192 + h * 32 + sel8);
      for (int tn = tn0; tn < tn0 + 8; ++tn) {
        v16h bb = *(const v16h*)(Klds + (tn * 16 + lanem) * 192 + h * 32 + sel * 16);
        v8f z = {0, 0, 0, 0, 0, 0, 0, 0};
        v8f d = wmma_f16(a, bb, z);
#pragma unroll
        for (int g = 0; g < 8; ++g)
          Sbuf[(tm * 16 + sel8 + g) * 256 + tn * 16 + lanem] = d[g];
      }
    }
    __syncthreads();

    // row softmax: 4 threads per row, quad reduction via shfl_xor (wave32)
    {
      int r  = tid >> 2;         // 0..63
      int qp = tid & 3;          // quarter of the 256-wide row
      const float* srow = Sbuf + r * 256 + qp * 64;
      _Float16* prow = Pbuf + r * 256 + qp * 64;
      float m = -3.0e38f;
      for (int j = 0; j < 64; ++j) m = fmaxf(m, srow[j]);
      m = fmaxf(m, __shfl_xor(m, 1, 32));
      m = fmaxf(m, __shfl_xor(m, 2, 32));
      float ssum = 0.f;
      for (int j = 0; j < 64; ++j) {
        float e = __expf(srow[j] - m);
        ssum += e;
        prow[j] = (_Float16)e;
      }
      ssum += __shfl_xor(ssum, 1, 32);
      ssum += __shfl_xor(ssum, 2, 32);
      float inv = 1.f / ssum;
      for (int j = 0; j < 64; ++j)
        prow[j] = (_Float16)((float)prow[j] * inv);
    }
    __syncthreads();

    // O_h = P @ V_h  (64x32, K=256) : 8 tiles, one per wave
    {
      int tm = wave >> 1, tn = wave & 1;
      v8f acc = {0, 0, 0, 0, 0, 0, 0, 0};
#pragma unroll
      for (int ks = 0; ks < 8; ++ks) {
        v16h a  = mkA(Pbuf + (tm * 16 + lanem) * 256 + ks * 32 + sel8);
        v16h bb = *(const v16h*)(Vt + (h * 32 + tn * 16 + lanem) * 256 + ks * 32 + sel * 16);
        acc = wmma_f16(a, bb, acc);
      }
#pragma unroll
      for (int g = 0; g < 8; ++g)
        Obuf[(tm * 16 + sel8 + g) * 32 + tn * 16 + lanem] = (_Float16)acc[g];
    }
    __syncthreads();

    // out += O_h @ Wo[h*32:(h+1)*32, :]  -- persistent accumulators
    // wave -> (tm = wave>>1, 6 N-tiles); one A-fragment per head.
    {
      int tm  = wave >> 1;
      int tn0 = (wave & 1) * 6;
      v16h a = mkA(Obuf + (tm * 16 + lanem) * 32 + sel8);
#pragma unroll
      for (int i = 0; i < 6; ++i) {
        int tn = tn0 + i;
        v16h bb = *(const v16h*)(WoT + (tn * 16 + lanem) * 192 + h * 32 + sel * 16);
        oacc[i] = wmma_f16(a, bb, oacc[i]);
      }
    }
    __syncthreads();
  }

  // ---------------- Stage E: transpose via LDS, +bo +residual, store -------
  {
    int tm  = wave >> 1;
    int tn0 = (wave & 1) * 6;
#pragma unroll
    for (int i = 0; i < 6; ++i) {
#pragma unroll
      for (int g = 0; g < 8; ++g)
        outb[(tm * 16 + sel8 + g) * 192 + (tn0 + i) * 16 + lanem] = oacc[i][g];
    }
  }
  __syncthreads();

  for (int s = tid; s < 192 * 8; s += 256) {
    int c = s >> 3, yy = s & 7;
    size_t base = ((size_t)b * 192 + c) * 65536 + (size_t)(wy * 8 + yy) * 256 + wx * 8;
    const float* rp = x + base;
    float* op = out + base;
    float bias = bo[c];
#pragma unroll
    for (int xx = 0; xx < 8; ++xx)
      op[xx] = outb[(yy * 8 + xx) * 192 + c] + bias + rp[xx];
  }
}

extern "C" void kernel_launch(void* const* d_in, const int* in_sizes, int n_in,
                              void* d_out, int out_size, void* d_ws, size_t ws_size,
                              hipStream_t stream) {
  (void)in_sizes; (void)n_in; (void)out_size; (void)ws_size;
  const float* x     = (const float*)d_in[0];
  const float* gamma = (const float*)d_in[1];
  const float* beta  = (const float*)d_in[2];
  const float* Wq    = (const float*)d_in[3];
  const float* bq    = (const float*)d_in[4];
  const float* Wkv   = (const float*)d_in[5];
  const float* bkv   = (const float*)d_in[6];
  const float* Wo    = (const float*)d_in[7];
  const float* bo    = (const float*)d_in[8];

  _Float16* WqT  = (_Float16*)d_ws;                       // 192*192 f16
  _Float16* WkvT = (_Float16*)((char*)d_ws + 73728);      // 384*192 f16
  _Float16* WoT  = (_Float16*)((char*)d_ws + 221184);     // 192*192 f16

  prep_weights<<<288, 256, 0, stream>>>(Wq, Wkv, Wo, WqT, WkvT, WoT);

  const int lds_bytes = 323584;  // 316 KB of the 320 KB WGP LDS
  (void)hipFuncSetAttribute((const void*)oca_main,
                            hipFuncAttributeMaxDynamicSharedMemorySize, lds_bytes);
  oca_main<<<4096, 256, lds_bytes, stream>>>(x, gamma, beta, bq, bkv, bo,
                                             WqT, WkvT, WoT, (float*)d_out);
}
